// IntraGraphAttention_67997922230509
// MI455X (gfx1250) — matrix operand order repaired and checked by reading, not compiled, not run on hardware
//
#include <hip/hip_runtime.h>

#define B_    8
#define L_    256
#define DN_   256
#define DE_   128
#define H_    8
#define DK_   32
#define DEH_  16
#define SLOPE_ 0.2f

typedef __bf16 bf16_t;
typedef __attribute__((ext_vector_type(16))) __bf16 v16bf;
typedef __attribute__((ext_vector_type(8)))  float  v8f;

// workspace offsets (in floats)
#define WS_NODE_PROJ 0u         // 524288
#define WS_NODE_WH   524288u    // 524288
#define WS_SI        1048576u   // 16384
#define WS_SJ        1064960u   // 16384
#define WS_TI        1081344u   // 262144
#define WS_TJC       1343488u   // 262144
#define WS_C         1605632u   // 128
#define WS_SEBIAS    1605760u   // 8
#define WS_VB        1605768u   // bf16 x 2048  (1024 floats)
#define WS_WEBF      1606792u   // bf16 x 16384 (8192 floats)
#define WS_SE        1614984u   // 4194304
// total 5809288 floats = 23.2 MB

__device__ __forceinline__ float lrelu_f(float x) { return x >= 0.f ? x : SLOPE_ * x; }

// A-fragment (16x32 bf16): lanes 0-15 = rows, hold K {0..15};
// lanes 16-31 = same rows, hold K {16..31}. Elements packed 2/VGPR in order.
__device__ __forceinline__ v16bf load_a_frag(const float4* a4, int kt, int hi) {
  int q = (kt * 32 + hi * 16) >> 2;
  float4 x0 = a4[q], x1 = a4[q + 1], x2 = a4[q + 2], x3 = a4[q + 3];
  v16bf av;
  av[0]=(bf16_t)x0.x; av[1]=(bf16_t)x0.y; av[2]=(bf16_t)x0.z; av[3]=(bf16_t)x0.w;
  av[4]=(bf16_t)x1.x; av[5]=(bf16_t)x1.y; av[6]=(bf16_t)x1.z; av[7]=(bf16_t)x1.w;
  av[8]=(bf16_t)x2.x; av[9]=(bf16_t)x2.y; av[10]=(bf16_t)x2.z; av[11]=(bf16_t)x2.w;
  av[12]=(bf16_t)x3.x; av[13]=(bf16_t)x3.y; av[14]=(bf16_t)x3.z; av[15]=(bf16_t)x3.w;
  return av;
}

// ---------------------------------------------------------------------------
// k0: fold weights.  W_E = ope_w @ blockdiag(Ew_e) @ We2_w  (bf16, n-major)
//     V[h,:] = w_e @ We_w[h*16:(h+1)*16, :]                  (bf16, padded to 16 rows)
//     C[o]   = ope_w @ (blockdiag(Ew_e) @ We2_b + tile(edge_b)) + ope_b
__global__ void __launch_bounds__(256)
k0_combine(const float* __restrict__ We_w,  const float* __restrict__ We_b,
           const float* __restrict__ We2_w, const float* __restrict__ We2_b,
           const float* __restrict__ attn_w, const float* __restrict__ edge_w,
           const float* __restrict__ edge_b, const float* __restrict__ ope_w,
           const float* __restrict__ ope_b,
           float* __restrict__ se_bias, float* __restrict__ Cvec,
           bf16_t* __restrict__ VB, bf16_t* __restrict__ WEbf) {
  __shared__ bf16_t M1[128 * 128];  // blockdiag(Ew_e) @ We2_w
  int t = threadIdx.x;
  if (t < 128) {
    int c = t;
    for (int h = 0; h < H_; ++h) {
      float s = 0.f;
      for (int d = 0; d < DEH_; ++d) s += attn_w[2 * DK_ + d] * We_w[(h * 16 + d) * 128 + c];
      VB[h * 128 + c] = (bf16_t)s;
      VB[(h + 8) * 128 + c] = (bf16_t)0.0f;  // zero padding rows 8..15
    }
  } else if (t < 136) {
    int h = t - 128;
    float s = 0.f;
    for (int d = 0; d < DEH_; ++d) s += We_b[h * 16 + d] * attn_w[2 * DK_ + d];
    se_bias[h] = s;
  }
  for (int q = 0; q < 64; ++q) {
    int idx = t * 64 + q;
    int r = idx >> 7, c = idx & 127;
    int h = r >> 4, a2 = r & 15;
    float s = 0.f;
    for (int d = 0; d < DEH_; ++d) s += edge_w[a2 * 80 + d] * We2_w[(h * 16 + d) * 128 + c];
    M1[r * 128 + c] = (bf16_t)s;
  }
  __syncthreads();
  int o = t & 127, half = t >> 7;
  for (int c = half * 64; c < half * 64 + 64; ++c) {
    float s = 0.f;
    for (int r = 0; r < 128; ++r) s += ope_w[o * 128 + r] * (float)M1[r * 128 + c];
    WEbf[o * 128 + c] = (bf16_t)s;  // n-major: [out_channel][k]
  }
  if (half == 0) {
    float s = ope_b[o];
    for (int r = 0; r < 128; ++r) {
      int h = r >> 4, a2 = r & 15;
      float g = edge_b[a2];
      for (int d = 0; d < DEH_; ++d) g += edge_w[a2 * 80 + d] * We2_b[h * 16 + d];
      s += ope_w[o * 128 + r] * g;
    }
    Cvec[o] = s;
  }
}

// ---------------------------------------------------------------------------
// k1a: node_proj = node @ Wn_w.T + Wn_b   (8 rows per block, f32)
__global__ void __launch_bounds__(256)
k1a_nodeproj(const float* __restrict__ node, const float* __restrict__ Wn_w,
             const float* __restrict__ Wn_b, float* __restrict__ node_proj) {
  __shared__ float xr[8 * DN_];
  int t = threadIdx.x;
  int r0 = blockIdx.x * 8;
  for (int idx = t; idx < 8 * DN_; idx += 256) xr[idx] = node[(size_t)r0 * DN_ + idx];
  __syncthreads();
  float acc[8];
  float bv = Wn_b[t];
#pragma unroll
  for (int k = 0; k < 8; ++k) acc[k] = bv;
  const float* w = Wn_w + (size_t)t * DN_;
  for (int c = 0; c < DN_; ++c) {
    float wv = w[c];
#pragma unroll
    for (int k = 0; k < 8; ++k) acc[k] += xr[k * DN_ + c] * wv;
  }
#pragma unroll
  for (int k = 0; k < 8; ++k) node_proj[(size_t)(r0 + k) * DN_ + t] = acc[k];
}

// k1b: node_wh = per-head node_proj @ Wh_w.T + Wh_b ; si/sj = head dots with attn_w
__global__ void __launch_bounds__(256)
k1b_wh_scores(const float* __restrict__ node_proj,
              const float* __restrict__ Wh_w, const float* __restrict__ Wh_b,
              const float* __restrict__ attn_w,
              float* __restrict__ node_wh, float* __restrict__ si_ws,
              float* __restrict__ sj_ws) {
  __shared__ float xp[8 * DN_];
  int t = threadIdx.x;
  int r0 = blockIdx.x * 8;
  for (int idx = t; idx < 8 * DN_; idx += 256) xp[idx] = node_proj[(size_t)r0 * DN_ + idx];
  __syncthreads();
  int h = t >> 5, d = t & 31;
  float acc[8];
  float bv = Wh_b[d];
#pragma unroll
  for (int k = 0; k < 8; ++k) acc[k] = bv;
  const float* w = Wh_w + d * DK_;
  for (int c = 0; c < DK_; ++c) {
    float wv = w[c];
#pragma unroll
    for (int k = 0; k < 8; ++k) acc[k] += xp[k * DN_ + h * DK_ + c] * wv;
  }
#pragma unroll
  for (int k = 0; k < 8; ++k) node_wh[(size_t)(r0 + k) * DN_ + t] = acc[k];
  if (t < 128) {
    int k = t >> 4, rest = t & 15, hh = rest & 7, which = rest >> 3;
    const float* aw = attn_w + which * DK_;
    float s = 0.f;
    for (int c = 0; c < DK_; ++c) s += xp[k * DN_ + hh * DK_ + c] * aw[c];
    (which ? sj_ws : si_ws)[(size_t)(r0 + k) * H_ + hh] = s;
  }
}

// ---------------------------------------------------------------------------
// k2: se = edge @ V.T  (bf16 WMMA, N=16 padded, 1 wave = 16-row tile)
__global__ void __launch_bounds__(256)
k2_se(const float* __restrict__ edge, const bf16_t* __restrict__ VB,
      const float* __restrict__ se_bias, float* __restrict__ se_out) {
  int t = threadIdx.x, wave = t >> 5, lane = t & 31, hi = lane >> 4, ln = lane & 15;
  int tile = blockIdx.x * 8 + wave;
  int R0 = tile * 16;
  const float4* a4 = (const float4*)(edge + (size_t)(R0 + ln) * DE_);
  float sb = (ln < 8) ? se_bias[ln] : 0.f;
  v8f c;
#pragma unroll
  for (int v = 0; v < 8; ++v) c[v] = sb;
#pragma unroll
  for (int kt = 0; kt < 4; ++kt) {
    v16bf a = load_a_frag(a4, kt, hi);
    union { uint4 u[2]; v16bf v; } bb;
    const uint4* bp = (const uint4*)(VB + ((size_t)ln * DE_ + kt * 32 + hi * 16));
    bb.u[0] = bp[0]; bb.u[1] = bp[1];
    c = __builtin_amdgcn_wmma_f32_16x16x32_bf16(false, a, false, bb.v, (short)0, c, false, false);
  }
  if (ln < 8) {
#pragma unroll
    for (int v = 0; v < 8; ++v)
      se_out[(size_t)(R0 + v + hi * 8) * H_ + ln] = c[v];
  }
}

// ---------------------------------------------------------------------------
// k3: per-(b,i): masked lrelu scores, softmax over j, aggregation, node output,
//     and Ti / (Tj + C) bias vectors for the final edge GEMM.
__global__ void __launch_bounds__(256)
k3_attn(const float* __restrict__ node_proj, const float* __restrict__ node_wh,
        const float* __restrict__ si_ws, const float* __restrict__ sj_ws,
        const float* __restrict__ se_ws,
        const float* __restrict__ opn_w, const float* __restrict__ opn_b,
        const float* __restrict__ Wn2_w, const float* __restrict__ Wn2_b,
        const float* __restrict__ edge_w, const float* __restrict__ ope_w,
        const float* __restrict__ Cvec,
        float* __restrict__ out_node, float* __restrict__ Ti_ws,
        float* __restrict__ Tjc_ws) {
  __shared__ float sc[L_ * H_];
  __shared__ float sii[H_], maxh[H_], sumh[H_];
  __shared__ float pre[DN_], nn[DN_], np2s[DN_], tp[256];
  int t = threadIdx.x;
  int bi = blockIdx.x;
  int b = bi >> 8, i = bi & 255;
  if (t < H_) sii[t] = si_ws[(size_t)bi * H_ + t];
  __syncthreads();
  {
    const float* sjrow = sj_ws + (size_t)b * L_ * H_;
    const float* serow = se_ws + (size_t)bi * L_ * H_;
    int j = t;
#pragma unroll
    for (int h = 0; h < H_; ++h) {
      float s = sii[h] + sjrow[j * H_ + h] + serow[j * H_ + h];
      if (j == i) s = -__builtin_inff();
      sc[j * H_ + h] = lrelu_f(s);
    }
  }
  __syncthreads();
  int h = t >> 5, lane = t & 31;
  {
    float m = -__builtin_inff();
    for (int j = lane; j < L_; j += 32) m = fmaxf(m, sc[j * H_ + h]);
    for (int off = 16; off; off >>= 1) m = fmaxf(m, __shfl_xor(m, off));
    if (lane == 0) maxh[h] = m;
  }
  __syncthreads();
  {
    float s = 0.f, mh = maxh[h];
    for (int j = lane; j < L_; j += 32) {
      float e = __expf(sc[j * H_ + h] - mh);
      sc[j * H_ + h] = e;
      s += e;
    }
    for (int off = 16; off; off >>= 1) s += __shfl_xor(s, off);
    if (lane == 0) sumh[h] = s;
  }
  __syncthreads();
  // aggregation: thread t -> (h, d), channel h*32+d == t
  float inv = 1.f / sumh[h];
  float acc = 0.f;
  const float* nw = node_wh + (size_t)b * L_ * DN_ + t;
  for (int j = 0; j < L_; ++j) acc += sc[j * H_ + h] * nw[(size_t)j * DN_];
  pre[t] = node_proj[(size_t)bi * DN_ + t] + lrelu_f(acc * inv);
  __syncthreads();
  float a1 = opn_b[t];
  {
    const float* w = opn_w + (size_t)t * DN_;
    for (int c = 0; c < DN_; ++c) a1 += pre[c] * w[c];
  }
  out_node[(size_t)bi * DN_ + t] = a1;
  nn[t] = a1;
  __syncthreads();
  float a2 = Wn2_b[t];
  {
    const float* w = Wn2_w + (size_t)t * DN_;
    for (int c = 0; c < DN_; ++c) a2 += nn[c] * w[c];
  }
  np2s[t] = a2;
  __syncthreads();
  {
    int which = t >> 7, r = t & 127, h2 = r >> 4, a_ = r & 15;
    const float* ew = edge_w + a_ * 80 + DEH_ + which * DK_;  // Ew_i / Ew_j
    float s = 0.f;
    for (int k = 0; k < DK_; ++k) s += np2s[h2 * DK_ + k] * ew[k];
    tp[t] = s;
  }
  __syncthreads();
  {
    int which = t >> 7, o = t & 127;
    const float* w2 = ope_w + (size_t)o * DE_;
    const float* src = tp + which * DE_;
    float s = 0.f;
    for (int r = 0; r < DE_; ++r) s += src[r] * w2[r];
    if (which == 0) Ti_ws[(size_t)bi * DE_ + o] = s;
    else            Tjc_ws[(size_t)bi * DE_ + o] = s + Cvec[o];
  }
}

// ---------------------------------------------------------------------------
// k4: new_edge = edge @ W_E.T + Ti[i] + Tjc[j]   (bf16 WMMA, 1 wave = 16x128 tile)
__global__ void __launch_bounds__(256)
k4_edge(const float* __restrict__ edge, const bf16_t* __restrict__ WEbf,
        const float* __restrict__ Ti, const float* __restrict__ Tjc,
        float* __restrict__ out_edge) {
  __shared__ bf16_t lw[128 * 128];  // 32 KB: W_E in bf16, n-major
  int t = threadIdx.x;
  {
    const uint4* src = (const uint4*)WEbf;
    uint4* dst = (uint4*)lw;
    for (int idx = t; idx < 2048; idx += 256) dst[idx] = src[idx];
  }
  __syncthreads();
  int wave = t >> 5, lane = t & 31, hi = lane >> 4, ln = lane & 15;
  int tile = blockIdx.x * 8 + wave;
  int R0 = tile * 16;
  int bi = R0 >> 8;          // b*256 + i   (16 consecutive j share (b,i))
  int b = R0 >> 16;
  int j0 = R0 & 255;
  const float4* a4 = (const float4*)(edge + (size_t)(R0 + ln) * DE_);
  v16bf A[4];
#pragma unroll
  for (int kt = 0; kt < 4; ++kt) A[kt] = load_a_frag(a4, kt, hi);
  const float* TiRow = Ti + (size_t)bi * DE_;
  const float* TjBase = Tjc + (size_t)(b * 256 + j0) * DE_;
#pragma unroll
  for (int nt = 0; nt < 8; ++nt) {
    int col = nt * 16 + ln;
    float tival = TiRow[col];
    v8f c;
#pragma unroll
    for (int v = 0; v < 8; ++v) c[v] = tival + TjBase[(size_t)(v + hi * 8) * DE_ + col];
#pragma unroll
    for (int kt = 0; kt < 4; ++kt) {
      union { uint4 u[2]; v16bf v; } bb;
      const uint4* bp = (const uint4*)(lw + ((size_t)(nt * 16 + ln) * 128 + kt * 32 + hi * 16));
      bb.u[0] = bp[0]; bb.u[1] = bp[1];
      c = __builtin_amdgcn_wmma_f32_16x16x32_bf16(false, A[kt], false, bb.v, (short)0, c, false, false);
    }
    float* orow = out_edge + (size_t)R0 * DE_ + col;
#pragma unroll
    for (int v = 0; v < 8; ++v) orow[(size_t)(v + hi * 8) * DE_] = c[v];
  }
}

// ---------------------------------------------------------------------------
extern "C" void kernel_launch(void* const* d_in, const int* in_sizes, int n_in,
                              void* d_out, int out_size, void* d_ws, size_t ws_size,
                              hipStream_t stream) {
  const float* node   = (const float*)d_in[0];
  const float* edge   = (const float*)d_in[1];
  const float* Wn_w   = (const float*)d_in[2];
  const float* Wn_b   = (const float*)d_in[3];
  const float* Wh_w   = (const float*)d_in[4];
  const float* Wh_b   = (const float*)d_in[5];
  const float* We_w   = (const float*)d_in[6];
  const float* We_b   = (const float*)d_in[7];
  const float* Wn2_w  = (const float*)d_in[8];
  const float* Wn2_b  = (const float*)d_in[9];
  const float* We2_w  = (const float*)d_in[10];
  const float* We2_b  = (const float*)d_in[11];
  const float* attn_w = (const float*)d_in[12];
  const float* edge_w = (const float*)d_in[13];
  const float* edge_b = (const float*)d_in[14];
  const float* opn_w  = (const float*)d_in[15];
  const float* opn_b  = (const float*)d_in[16];
  const float* ope_w  = (const float*)d_in[17];
  const float* ope_b  = (const float*)d_in[18];
  (void)in_sizes; (void)n_in; (void)out_size; (void)ws_size;

  float* out = (float*)d_out;
  float* ws  = (float*)d_ws;

  float*  node_proj = ws + WS_NODE_PROJ;
  float*  node_wh   = ws + WS_NODE_WH;
  float*  si        = ws + WS_SI;
  float*  sj        = ws + WS_SJ;
  float*  Ti        = ws + WS_TI;
  float*  Tjc       = ws + WS_TJC;
  float*  Cvec      = ws + WS_C;
  float*  se_bias   = ws + WS_SEBIAS;
  bf16_t* VB        = (bf16_t*)(ws + WS_VB);
  bf16_t* WEbf      = (bf16_t*)(ws + WS_WEBF);
  float*  se        = ws + WS_SE;

  k0_combine<<<1, 256, 0, stream>>>(We_w, We_b, We2_w, We2_b, attn_w, edge_w,
                                    edge_b, ope_w, ope_b, se_bias, Cvec, VB, WEbf);
  k1a_nodeproj<<<256, 256, 0, stream>>>(node, Wn_w, Wn_b, node_proj);
  k1b_wh_scores<<<256, 256, 0, stream>>>(node_proj, Wh_w, Wh_b, attn_w,
                                         node_wh, si, sj);
  k2_se<<<4096, 256, 0, stream>>>(edge, VB, se_bias, se);
  k3_attn<<<2048, 256, 0, stream>>>(node_proj, node_wh, si, sj, se,
                                    opn_w, opn_b, Wn2_w, Wn2_b, edge_w, ope_w,
                                    Cvec, out, Ti, Tjc);
  k4_edge<<<4096, 256, 0, stream>>>(edge, WEbf, Ti, Tjc, out + (size_t)B_ * L_ * DN_);
}
